// SelfAttention_49752901157332
// MI455X (gfx1250) — matrix operand order repaired and checked by reading, not compiled, not run on hardware
//
#include <hip/hip_runtime.h>
#include <hip/hip_bf16.h>

// ---------------------------------------------------------------------------
// Causal self-attention for MI455X (gfx1250, wave32, WMMA bf16 16x16x32).
//   B=4, T=4096, D_MODEL=1024, HEAD_DIM=64
// Round 2: software-pipelined loads (double-buffered fragments) so WMMAs
// overlap VMEM instead of the load->s_wait_loadcnt(0)->wmma serialization
// seen in round 1 assembly.
// ---------------------------------------------------------------------------

#define DM 1024
#define HD 64
#define TT 4096
#define NBATCH 4
#define BT (NBATCH * TT)

typedef __attribute__((ext_vector_type(16))) __bf16 v16bf;
typedef __attribute__((ext_vector_type(8)))  float  v8f;

union Frag16 {
    v16bf  v;
    __bf16 h[16];
    uint4  u[2];
};

__device__ __forceinline__ v8f wmma_bf16(const v16bf& a, const v16bf& b, const v8f& c) {
    // D = A(16x32 bf16) x B(32x16 bf16) + C(16x16 f32)
    return __builtin_amdgcn_wmma_f32_16x16x32_bf16(
        /*neg_a=*/false, a, /*neg_b=*/false, b,
        /*c_mod=*/(short)0, c, /*reuse_a=*/false, /*reuse_b=*/false);
}

__device__ __forceinline__ v8f v8f_zero() {
    v8f z = {0.f, 0.f, 0.f, 0.f, 0.f, 0.f, 0.f, 0.f};
    return z;
}

// ---------------------------------------------------------------------------
// Kernel 0: transpose + convert weights. wt holds 3 matrices [3][64][1024].
// ---------------------------------------------------------------------------
__global__ __launch_bounds__(256)
void k_prep_w(const float* __restrict__ Wq, const float* __restrict__ Wk,
              const float* __restrict__ Wv, __bf16* __restrict__ wt) {
    int idx = blockIdx.x * 256 + threadIdx.x;      // 3 * 64 * 1024 threads
    int mat = idx >> 16;
    int r   = idx & 0xFFFF;
    int k   = r >> 6;
    int n   = r & 63;
    const float* src = (mat == 0) ? Wq : (mat == 1) ? Wk : Wv;
    wt[(size_t)mat * (HD * DM) + (size_t)n * DM + k] = (__bf16)src[k * HD + n];
}

// ---------------------------------------------------------------------------
// Kernel 1: fused QKV projection. One wave = 16 rows x 64 cols x {q,k,v}.
// Double-buffered A (x) and B (weight) fragments for load/WMMA overlap.
// ---------------------------------------------------------------------------
__global__ __launch_bounds__(256)
void k_qkv(const float* __restrict__ x,
           const __bf16* __restrict__ wt,   // [3][64][1024] bf16, n-major
           const float* __restrict__ bq, const float* __restrict__ bk,
           const float* __restrict__ bv,
           __bf16* __restrict__ qws,        // [BT][64]
           __bf16* __restrict__ kws,        // [BT][64]
           __bf16* __restrict__ vws) {      // [B][64][T] (transposed)
    const int lane = threadIdx.x & 31;
    const int half = lane >> 4;
    const int l16  = lane & 15;
    const int wave = (blockIdx.x * 256 + threadIdx.x) >> 5;
    const int rb   = wave * 16;                        // first of 16 rows
    const float* xrow = x + (size_t)(rb + l16) * DM;   // A-frag row for this lane

    // Load the three B-fragments (q,k,v weights) for N-tile `tt`, K-offset `wkk`.
    auto loadB = [&](Frag16* d, int tt, int wkk) {
        const __bf16* wp = wt + (size_t)(tt * 16 + l16) * DM + wkk;
        d[0].u[0] = *(const uint4*)(wp);                 d[0].u[1] = *(const uint4*)(wp + 8);
        d[1].u[0] = *(const uint4*)(wp + HD * DM);       d[1].u[1] = *(const uint4*)(wp + HD * DM + 8);
        d[2].u[0] = *(const uint4*)(wp + 2 * HD * DM);   d[2].u[1] = *(const uint4*)(wp + 2 * HD * DM + 8);
    };
    auto loadX = [&](float4* f, int kk) {
        const int kb = kk * 32 + half * 16;
        f[0] = *(const float4*)(xrow + kb);
        f[1] = *(const float4*)(xrow + kb + 4);
        f[2] = *(const float4*)(xrow + kb + 8);
        f[3] = *(const float4*)(xrow + kb + 12);
    };

    v8f cq[4], ck[4], cv[4];
#pragma unroll
    for (int t = 0; t < 4; ++t) { cq[t] = v8f_zero(); ck[t] = v8f_zero(); cv[t] = v8f_zero(); }

    float4 xc[4], xn[4];
    Frag16 bc[3], bn[3];
    loadX(xc, 0);
    loadB(bc, 0, half * 16);                           // kk=0, tile 0

#pragma unroll 1
    for (int kk = 0; kk < DM / 32; ++kk) {
        // A fragment: fp32 -> bf16 (K contiguous half*16..half*16+15)
        Frag16 a;
        a.h[0] = (__bf16)xc[0].x; a.h[1] = (__bf16)xc[0].y; a.h[2]  = (__bf16)xc[0].z; a.h[3]  = (__bf16)xc[0].w;
        a.h[4] = (__bf16)xc[1].x; a.h[5] = (__bf16)xc[1].y; a.h[6]  = (__bf16)xc[1].z; a.h[7]  = (__bf16)xc[1].w;
        a.h[8] = (__bf16)xc[2].x; a.h[9] = (__bf16)xc[2].y; a.h[10] = (__bf16)xc[2].z; a.h[11] = (__bf16)xc[2].w;
        a.h[12]= (__bf16)xc[3].x; a.h[13]= (__bf16)xc[3].y; a.h[14] = (__bf16)xc[3].z; a.h[15] = (__bf16)xc[3].w;

        const int kkn = (kk + 1 < DM / 32) ? kk + 1 : kk;
        loadX(xn, kkn);                                // prefetch next K-step of x

        const int wk  = kk  * 32 + half * 16;
        const int wkn = kkn * 32 + half * 16;
#pragma unroll
        for (int t = 0; t < 4; ++t) {
            // prefetch B-fragments for next tile (or tile 0 of next K-step)
            if (t < 3) loadB(bn, t + 1, wk);
            else       loadB(bn, 0, wkn);
            cq[t] = wmma_bf16(a.v, bc[0].v, cq[t]);
            ck[t] = wmma_bf16(a.v, bc[1].v, ck[t]);
            cv[t] = wmma_bf16(a.v, bc[2].v, cv[t]);
            bc[0] = bn[0]; bc[1] = bn[1]; bc[2] = bn[2];
        }
#pragma unroll
        for (int i = 0; i < 4; ++i) xc[i] = xn[i];
    }

    // ---- bias + store (C layout: row = half*8 + v, col = t*16 + l16)
#pragma unroll
    for (int t = 0; t < 4; ++t) {
        const int ncol = t * 16 + l16;
        const float bqv = bq[ncol], bkv = bk[ncol], bvv = bv[ncol];
#pragma unroll
        for (int v = 0; v < 8; ++v) {
            const int row = rb + half * 8 + v;
            qws[(size_t)row * HD + ncol] = (__bf16)(cq[t][v] + bqv);
            kws[(size_t)row * HD + ncol] = (__bf16)(ck[t][v] + bkv);
            const int bi = row >> 12;                  // row / 4096
            const int tt = row & (TT - 1);
            vws[((size_t)bi * HD + ncol) * TT + tt] = (__bf16)(cv[t][v] + bvv);
        }
    }
}

// ---------------------------------------------------------------------------
// Kernel 2: causal flash attention. One wave = 16 query rows; 32-key blocks.
// K-fragments double-buffered across blocks; V loads issued before the
// softmax/LDS fence so they overlap the VALU work.
// ---------------------------------------------------------------------------
__global__ __launch_bounds__(256)
void k_attn(const __bf16* __restrict__ qws, const __bf16* __restrict__ kws,
            const __bf16* __restrict__ vws, float* __restrict__ out) {
    __shared__ __bf16 ldsP[8 * 16 * 32];               // 1 KB P-tile per wave
    const int lane = threadIdx.x & 31;
    const int half = lane >> 4;
    const int l16  = lane & 15;
    const int wid  = threadIdx.x >> 5;
    __bf16* pTile  = ldsP + wid * (16 * 32);

    const int wave = blockIdx.x * 8 + wid;
    const int b    = wave >> 8;                        // 256 q-tiles per batch
    const int qt   = wave & 255;
    const int rb   = qt * 16;                          // tile's first query row
    const int rbg  = b * TT + rb;

    // ---- Q fragments (A layout), loaded once.
    Frag16 q0, q1;
    {
        const __bf16* qrow = qws + (size_t)(rbg + l16) * HD;
        q0.u[0] = *(const uint4*)(qrow + half * 16);
        q0.u[1] = *(const uint4*)(qrow + half * 16 + 8);
        q1.u[0] = *(const uint4*)(qrow + 32 + half * 16);
        q1.u[1] = *(const uint4*)(qrow + 32 + half * 16 + 8);
    }

    // K B-fragments for key block kb: per-lane 32B contiguous loads.
    auto loadK = [&](Frag16* d, int kb) {
        const int keyg = b * TT + kb * 32;
        const __bf16* kr0 = kws + (size_t)(keyg + l16) * HD;
        const __bf16* kr1 = kws + (size_t)(keyg + 16 + l16) * HD;
        d[0].u[0] = *(const uint4*)(kr0 + half * 16);      d[0].u[1] = *(const uint4*)(kr0 + half * 16 + 8);
        d[1].u[0] = *(const uint4*)(kr0 + 32 + half * 16); d[1].u[1] = *(const uint4*)(kr0 + 32 + half * 16 + 8);
        d[2].u[0] = *(const uint4*)(kr1 + half * 16);      d[2].u[1] = *(const uint4*)(kr1 + half * 16 + 8);
        d[3].u[0] = *(const uint4*)(kr1 + 32 + half * 16); d[3].u[1] = *(const uint4*)(kr1 + 32 + half * 16 + 8);
    };

    v8f o[4];
#pragma unroll
    for (int t = 0; t < 4; ++t) o[t] = v8f_zero();
    float m[8], l[8];
#pragma unroll
    for (int v = 0; v < 8; ++v) { m[v] = -__builtin_inff(); l[v] = 0.f; }

    const float scale = 0.125f;                        // 1/sqrt(64)
    const int nkb = (rb >> 5) + 1;                     // causal key-block count

    Frag16 kc[4], kn[4];
    loadK(kc, 0);

#pragma unroll 1
    for (int kb = 0; kb < nkb; ++kb) {
        // ---- V B-fragments (vT layout: h fixed per lane, keys contiguous).
        // Issued first so they overlap the softmax VALU work below.
        Frag16 vf[4];
#pragma unroll
        for (int t = 0; t < 4; ++t) {
            const __bf16* vp = vws + ((size_t)(b * HD + t * 16 + l16)) * TT + kb * 32 + half * 16;
            vf[t].u[0] = *(const uint4*)(vp);
            vf[t].u[1] = *(const uint4*)(vp + 8);
        }

        // ---- S = Q K^T from the pre-loaded K fragments.
        v8f s0 = wmma_bf16(q0.v, kc[0].v, v8f_zero());
        s0     = wmma_bf16(q1.v, kc[1].v, s0);
        v8f s1 = wmma_bf16(q0.v, kc[2].v, v8f_zero());
        s1     = wmma_bf16(q1.v, kc[3].v, s1);

        // ---- prefetch next block's K fragments (overlaps softmax + PV).
        loadK(kn, (kb + 1 < nkb) ? kb + 1 : kb);

        // ---- causal mask + online softmax (row = half*8+v, col = lane)
        float alpha[8];
#pragma unroll
        for (int v = 0; v < 8; ++v) {
            const int row  = rb + half * 8 + v;
            const int key0 = kb * 32 + l16;
            float a0 = (key0      <= row) ? s0[v] * scale : -__builtin_inff();
            float a1 = (key0 + 16 <= row) ? s1[v] * scale : -__builtin_inff();
            float t = fmaxf(a0, a1);                   // row-max over 16 lanes
            t = fmaxf(t, __shfl_xor(t, 1, 16));
            t = fmaxf(t, __shfl_xor(t, 2, 16));
            t = fmaxf(t, __shfl_xor(t, 4, 16));
            t = fmaxf(t, __shfl_xor(t, 8, 16));
            const float mn = fmaxf(m[v], t);
            a0 = __expf(a0 - mn);
            a1 = __expf(a1 - mn);
            float rs = a0 + a1;                        // row-sum over 16 lanes
            rs += __shfl_xor(rs, 1, 16);
            rs += __shfl_xor(rs, 2, 16);
            rs += __shfl_xor(rs, 4, 16);
            rs += __shfl_xor(rs, 8, 16);
            alpha[v] = __expf(m[v] - mn);
            l[v] = l[v] * alpha[v] + rs;
            m[v] = mn;
            // spill P (C layout) to the wave-private LDS tile as bf16
            pTile[(half * 8 + v) * 32 + l16]      = (__bf16)a0;
            pTile[(half * 8 + v) * 32 + l16 + 16] = (__bf16)a1;
        }

        // DS ops are in-order per wave; fence so the compiler keeps them so.
        asm volatile("s_wait_dscnt 0" ::: "memory");

        // ---- reload P in A layout (row = l16, K contiguous half*16..half*16+15)
        Frag16 pf;
        {
            const __bf16* pr = pTile + l16 * 32 + half * 16;
            pf.u[0] = *(const uint4*)(pr);
            pf.u[1] = *(const uint4*)(pr + 8);
        }

        // ---- rescale accumulators, then O += P V
#pragma unroll
        for (int t = 0; t < 4; ++t)
#pragma unroll
            for (int v = 0; v < 8; ++v) o[t][v] *= alpha[v];

#pragma unroll
        for (int t = 0; t < 4; ++t) o[t] = wmma_bf16(pf.v, vf[t].v, o[t]);

        // rotate K double-buffer
#pragma unroll
        for (int t = 0; t < 4; ++t) kc[t] = kn[t];
    }

    // ---- normalize and store fp32 output
#pragma unroll
    for (int t = 0; t < 4; ++t) {
        const int h = t * 16 + l16;
#pragma unroll
        for (int v = 0; v < 8; ++v) {
            const int row = rbg + half * 8 + v;
            out[(size_t)row * HD + h] = o[t][v] / l[v];
        }
    }
}

// ---------------------------------------------------------------------------
// Launch: ws layout = q(2MB) | k(2MB) | vT(2MB) | Wt bf16 (3 x 128KB)
// ---------------------------------------------------------------------------
extern "C" void kernel_launch(void* const* d_in, const int* in_sizes, int n_in,
                              void* d_out, int out_size, void* d_ws, size_t ws_size,
                              hipStream_t stream) {
    const float* x  = (const float*)d_in[0];
    const float* Wq = (const float*)d_in[1];
    const float* bq = (const float*)d_in[2];
    const float* Wk = (const float*)d_in[3];
    const float* bk = (const float*)d_in[4];
    const float* Wv = (const float*)d_in[5];
    const float* bv = (const float*)d_in[6];

    char* ws = (char*)d_ws;
    __bf16* qws = (__bf16*)(ws);
    __bf16* kws = (__bf16*)(ws + (size_t)2 * 1024 * 1024);
    __bf16* vws = (__bf16*)(ws + (size_t)4 * 1024 * 1024);
    __bf16* wt  = (__bf16*)(ws + (size_t)6 * 1024 * 1024);

    k_prep_w<<<768, 256, 0, stream>>>(Wq, Wk, Wv, wt);                 // 3*64*1024 elems
    k_qkv   <<<BT / 128, 256, 0, stream>>>(x, wt, bq, bk, bv, qws, kws, vws); // 128 blocks
    k_attn  <<<(NBATCH * TT / 16) / 8, 256, 0, stream>>>(qws, kws, vws, (float*)d_out);
}
